// AutoQVLALinearMixedW_56581899157847
// MI455X (gfx1250) — matrix operand order repaired and checked by reading, not compiled, not run on hardware
//
#include <hip/hip_runtime.h>

// ---- CDNA5 WMMA vector types ----
typedef __attribute__((ext_vector_type(16))) _Float16 v16h;
typedef __attribute__((ext_vector_type(8)))  float    v8f;
typedef __attribute__((ext_vector_type(4)))  float    f32x4;
typedef __attribute__((ext_vector_type(4)))  int      i32x4;
typedef __attribute__((ext_vector_type(2)))  unsigned int u32x2;

#define K_DIM 4096
#define OUT_N 11008

enum { W_F32 = 0, W_I32 = 1, W_P4 = 2 };

// -------- zero-fill output (pruned channels must stay 0; d_out is poisoned) --------
__global__ void zero_out_kernel(f32x4* __restrict__ out, int n4) {
    int i = blockIdx.x * blockDim.x + threadIdx.x;
    if (i < n4) {
        f32x4 z = {0.f, 0.f, 0.f, 0.f};
        out[i] = z;
    }
}

// -------- mixed-precision WMMA GEMM --------
// Each wave computes a 16(M) x 64(N) tile: 1 A-fragment reused by 4 WMMAs per K-step.
// A fragment (16x32 f16): lane l, h=l>>4, row=l&15,
//   element i -> K = kBase + 16h + i   (16 consecutive K per lane)
// B fragment (32x16 f16): lane l holds column nBase+16t+(l&15),
//   element i -> K = kBase + 16h + i   (16 consecutive K per lane)
// C/D (16x16 f32): VGPR j -> row mBase + j + 8h, column = lane's column.
template <int WT>
__global__ __launch_bounds__(256, 1) void mixed_wmma_gemm(
    const float* __restrict__ x,
    const void*  __restrict__ wraw,
    const float* __restrict__ scale,   // per-output-channel weight scale (unused for W_F32)
    const float* __restrict__ bias,
    const int*   __restrict__ idx,     // scatter indices into the 11008 output channels
    float*       __restrict__ out,
    int N)
{
    const int lane = threadIdx.x & 31;
    const int wave = threadIdx.x >> 5;
    const int l15  = lane & 15;
    const int h    = lane >> 4;

    const int nTiles = N >> 6;                   // number of 64-wide N tiles
    const int job    = blockIdx.x * 8 + wave;    // 8 waves per 256-thread block
    const int mBase  = (job / nTiles) << 4;
    const int nBase  = (job % nTiles) << 6;

    const float* aPtr = x + (size_t)(mBase + l15) * K_DIM + 16 * h;

    const float*       wf = (const float*)wraw;
    const int*         wi = (const int*)wraw;
    const signed char* wp = (const signed char*)wraw;

    int nrow[4];
#pragma unroll
    for (int t = 0; t < 4; ++t) nrow[t] = nBase + t * 16 + l15;

    v8f acc[4];
#pragma unroll
    for (int t = 0; t < 4; ++t)
#pragma unroll
        for (int j = 0; j < 8; ++j) acc[t][j] = 0.f;

    for (int k = 0; k < K_DIM; k += 32) {
        // ---- build A fragment (f32 -> f16) ----
        f32x4 a0 = *(const f32x4*)(aPtr + k);
        f32x4 a1 = *(const f32x4*)(aPtr + k + 4);
        f32x4 a2 = *(const f32x4*)(aPtr + k + 8);
        f32x4 a3 = *(const f32x4*)(aPtr + k + 12);
        v16h A;
#pragma unroll
        for (int i = 0; i < 4; ++i) {
            A[i]      = (_Float16)a0[i];
            A[i + 4]  = (_Float16)a1[i];
            A[i + 8]  = (_Float16)a2[i];
            A[i + 12] = (_Float16)a3[i];
        }

#pragma unroll
        for (int t = 0; t < 4; ++t) {
            v16h B;
            if (WT == W_F32) {
                const float* bp = wf + (size_t)nrow[t] * K_DIM + 16 * h + k;
                f32x4 b0 = *(const f32x4*)(bp);
                f32x4 b1 = *(const f32x4*)(bp + 4);
                f32x4 b2 = *(const f32x4*)(bp + 8);
                f32x4 b3 = *(const f32x4*)(bp + 12);
#pragma unroll
                for (int i = 0; i < 4; ++i) {
                    B[i]      = (_Float16)b0[i];
                    B[i + 4]  = (_Float16)b1[i];
                    B[i + 8]  = (_Float16)b2[i];
                    B[i + 12] = (_Float16)b3[i];
                }
            } else if (WT == W_I32) {
                const int* bp = wi + (size_t)nrow[t] * K_DIM + 16 * h + k;
                i32x4 b0 = *(const i32x4*)(bp);
                i32x4 b1 = *(const i32x4*)(bp + 4);
                i32x4 b2 = *(const i32x4*)(bp + 8);
                i32x4 b3 = *(const i32x4*)(bp + 12);
#pragma unroll
                for (int i = 0; i < 4; ++i) {
                    B[i]      = (_Float16)b0[i];   // int8 range: exact in f16
                    B[i + 4]  = (_Float16)b1[i];
                    B[i + 8]  = (_Float16)b2[i];
                    B[i + 12] = (_Float16)b3[i];
                }
            } else { // W_P4: two signed nibbles per byte, 8 bytes -> 16 f16
                const signed char* bp = wp + (size_t)nrow[t] * (K_DIM / 2) + 8 * h + (k >> 1);
                u32x2 pk = *(const u32x2*)bp;
#pragma unroll
                for (int d = 0; d < 2; ++d) {
                    unsigned int word = pk[d];
#pragma unroll
                    for (int j = 0; j < 4; ++j) {
                        int byte = (int)((word >> (8 * j)) & 0xFF);
                        int lo = (byte << 28) >> 28;   // sign-extend low nibble
                        int hi = (byte << 24) >> 28;   // sign-extend high nibble
                        B[d * 8 + j * 2]     = (_Float16)lo;
                        B[d * 8 + j * 2 + 1] = (_Float16)hi;
                    }
                }
            }
            acc[t] = __builtin_amdgcn_wmma_f32_16x16x32_f16(
                false, A, false, B, (short)0, acc[t], false, false);
        }
    }

    // ---- epilogue: per-channel scale + bias, scatter columns via idx ----
#pragma unroll
    for (int t = 0; t < 4; ++t) {
        const int   n  = nrow[t];
        const float s  = (WT == W_F32) ? 1.0f : scale[n];
        const float bb = bias[n];
        float* op = out + (size_t)(mBase + 8 * h) * OUT_N + idx[n];
#pragma unroll
        for (int j = 0; j < 8; ++j)
            op[(size_t)j * OUT_N] = acc[t][j] * s + bb;
    }
}

extern "C" void kernel_launch(void* const* d_in, const int* in_sizes, int n_in,
                              void* d_out, int out_size, void* d_ws, size_t ws_size,
                              hipStream_t stream) {
    (void)in_sizes; (void)n_in; (void)d_ws; (void)ws_size;

    const float*       x     = (const float*)d_in[0];
    const float*       w16   = (const float*)d_in[1];
    const float*       b16   = (const float*)d_in[2];
    const int*         q8    = (const int*)d_in[3];       // int32 storage
    const float*       s8    = (const float*)d_in[4];
    const float*       b8    = (const float*)d_in[5];
    const signed char* p4    = (const signed char*)d_in[6];
    const float*       s4    = (const float*)d_in[7];
    const float*       b4    = (const float*)d_in[8];
    const signed char* p2    = (const signed char*)d_in[9];
    const float*       s2    = (const float*)d_in[10];
    const float*       b2    = (const float*)d_in[11];
    const int*         idx16 = (const int*)d_in[12];
    const int*         idx8  = (const int*)d_in[13];
    const int*         idx4  = (const int*)d_in[14];
    const int*         idx2  = (const int*)d_in[15];
    float* out = (float*)d_out;

    // Zero the full output (pruned channels stay 0).
    const int n4 = out_size >> 2;
    zero_out_kernel<<<(n4 + 255) / 256, 256, 0, stream>>>((f32x4*)out, n4);

    // jobs = (M/16) * (N/64); 8 waves (jobs) per 256-thread block.
    const int M_TILES = 256 / 16;
    mixed_wmma_gemm<W_F32><<<(M_TILES * (1024 / 64)) / 8, 256, 0, stream>>>(
        x, w16, nullptr, b16, idx16, out, 1024);
    mixed_wmma_gemm<W_I32><<<(M_TILES * (3072 / 64)) / 8, 256, 0, stream>>>(
        x, q8, s8, b8, idx8, out, 3072);
    mixed_wmma_gemm<W_P4><<<(M_TILES * (4096 / 64)) / 8, 256, 0, stream>>>(
        x, p4, s4, b4, idx4, out, 4096);
    mixed_wmma_gemm<W_P4><<<(M_TILES * (2048 / 64)) / 8, 256, 0, stream>>>(
        x, p2, s2, b2, idx2, out, 2048);
}
